// CorrelationLayer_68487548502302
// MI455X (gfx1250) — matrix operand order repaired and checked by reading, not compiled, not run on hardware
//
#include <hip/hip_runtime.h>
#include <hip/hip_bf16.h>

// CDNA5 / gfx1250 fused correlation layer.
// corr[b,d,y,x] = <f1_hat[:,y,x], f2_hat[:,y+dy,x+dx]>, d = (dy+4)*9 + (dx+4)
// Normalization folded into output scaling; dots computed with
// v_wmma_f32_16x16x32_f16 (fp16 inputs, fp32 accumulate).

typedef _Float16 v16h __attribute__((ext_vector_type(16)));
typedef _Float16 v8h  __attribute__((ext_vector_type(8)));
typedef float    v8f  __attribute__((ext_vector_type(8)));

#define CCH   256          // channels (K dim), 8 WMMA k-steps of 32
#define HH    96
#define WW    128
#define ND    81           // (2*4+1)^2 displacements
#define XB    64           // x-block per workgroup (4 waves x 16 px)
#define F2W   80           // staged f2 width: XB + 2*MAX_DISP, padded to 2 N-tiles
#define STRD  264          // LDS half-stride per pixel: 256 + 8 pad (16B aligned)

__global__ __launch_bounds__(128)
void corr81_wmma_kernel(const float* __restrict__ f1,
                        const float* __restrict__ f2,
                        float* __restrict__ out) {
  __shared__ _Float16 buf[F2W * STRD];   // 42,240 B: f1 tile first, then f2 rows
  __shared__ float s_inv1[XB];
  __shared__ float s_inv2[F2W];
  __shared__ float s_red[128];
  __shared__ float s_out[9 * XB];

  const int t  = threadIdx.x;
  const int wg = blockIdx.x;
  const int x0 = (wg & 1) * XB;              // W / XB == 2
  const int y  = (wg >> 1) % HH;
  const int b  = wg / (2 * HH);

  // ---- Stage f1 tile (64 px x 256 ch) to LDS as fp16; fp32 sum-of-squares ----
  {
    const int xl   = t & 63;
    const int cpar = t >> 6;                 // threads split channels 2-way
    const float* p = f1 + ((size_t)b * CCH * HH + (size_t)y) * WW + x0 + xl;
    float ss = 0.f;
    #pragma unroll 4
    for (int i = 0; i < CCH / 2; ++i) {
      const int c = cpar + 2 * i;
      const float v = p[(size_t)c * (HH * WW)];   // coalesced: lanes -> consecutive x
      ss += v * v;
      buf[xl * STRD + c] = (_Float16)v;
    }
    s_red[t] = ss;
  }
  __syncthreads();
  if (t < XB)
    s_inv1[t] = 1.0f / fmaxf(sqrtf(s_red[t] + s_red[t + 64]), 1e-12f);
  __syncthreads();

  const int wv   = t >> 5;      // wave id: 16-px M tile
  const int lane = t & 31;
  const int n    = lane & 15;   // A row m / B-C-D column n
  const int kh   = lane >> 4;   // lane half selects K sub-block

  // ---- Hoist A fragments (dy-invariant) to registers per ISA 7.12.2 layout ----
  // A 16x32 f16: lane<16 holds K=[8kh..8kh+8) in v0-3 and K+16..K+24 in v4-7.
  v16h afr[8];
  {
    const int apx = 16 * wv + n;
    #pragma unroll
    for (int s = 0; s < 8; ++s) {
      const _Float16* p = &buf[apx * STRD + 32 * s + 8 * kh];
      const v8h lo = *(const v8h*)(p);        // ds_load_b128
      const v8h hi = *(const v8h*)(p + 16);   // ds_load_b128
      afr[s] = __builtin_shufflevector(lo, hi, 0,1,2,3,4,5,6,7,
                                               8,9,10,11,12,13,14,15);
    }
  }
  float inv1r[8];
  #pragma unroll
  for (int r = 0; r < 8; ++r) inv1r[r] = s_inv1[16 * wv + r + 8 * kh];
  __syncthreads();   // f1 reads done; buf is reused for f2 rows below

  const float* f2b = f2 + (size_t)b * CCH * HH * WW;

  for (int dyi = 0; dyi < 9; ++dyi) {
    const int  y2    = y + dyi - 4;
    const bool rowok = (y2 >= 0) && (y2 < HH);

    // ---- Stage f2 row segment (80 px, zero-padded) + inverse norms ----
    if (t < F2W) {
      const int  xg = x0 - 4 + t;
      const bool ok = rowok && (xg >= 0) && (xg < WW);
      const float* p = f2b + (size_t)y2 * WW + xg;
      float ss = 0.f;
      #pragma unroll 4
      for (int c = 0; c < CCH; ++c) {
        float v = 0.f;
        if (ok) v = p[(size_t)c * (HH * WW)];   // coalesced across lanes
        ss += v * v;
        buf[t * STRD + c] = (_Float16)v;
      }
      s_inv2[t] = 1.0f / fmaxf(sqrtf(ss), 1e-12f);
    }
    __syncthreads();

    // ---- Banded GEMM: two N=16 tiles cover dx in [-4,4] for this M tile ----
    #pragma unroll
    for (int j = 0; j < 2; ++j) {
      v8f acc = {};
      const int bpx = 16 * wv + 16 * j + n;   // f2 LDS pixel = B column n
      #pragma unroll
      for (int s = 0; s < 8; ++s) {
        // B 32x16 f16: lane half kh holds K=[8kh..8kh+8) then K=[16+8kh..24+8kh), matching A.
        const _Float16* p = &buf[bpx * STRD + 32 * s + 8 * kh];
        const v8h lo = *(const v8h*)(p);
        const v8h hi = *(const v8h*)(p + 16);
        const v16h bfr = __builtin_shufflevector(lo, hi, 0,1,2,3,4,5,6,7,
                                                         8,9,10,11,12,13,14,15);
        acc = __builtin_amdgcn_wmma_f32_16x16x32_f16(
                  false, afr[s], false, bfr, (short)0, acc, false, false);
      }
      // Band extraction: D vgpr r -> M = r + 8*kh, N = n; dx+4 = 16j + n - m.
      const float in2 = s_inv2[bpx];
      #pragma unroll
      for (int r = 0; r < 8; ++r) {
        const int m   = r + 8 * kh;
        const int dxi = 16 * j + n - m;
        if (dxi >= 0 && dxi <= 8)
          s_out[dxi * XB + 16 * wv + m] = acc[r] * inv1r[r] * in2;
      }
    }
    __syncthreads();

    // ---- Coalesced output: 64 consecutive floats per displacement plane ----
    for (int i = t; i < 9 * XB; i += 128) {
      const int dxi = i >> 6;
      const int xl  = i & 63;
      out[(((size_t)b * ND + (dyi * 9 + dxi)) * HH + y) * WW + x0 + xl] = s_out[i];
    }
    // next iteration's staging barrier orders s_out reads vs. next writes
  }
}

extern "C" void kernel_launch(void* const* d_in, const int* in_sizes, int n_in,
                              void* d_out, int out_size, void* d_ws, size_t ws_size,
                              hipStream_t stream) {
  const float* f1 = (const float*)d_in[0];
  const float* f2 = (const float*)d_in[1];
  float* out = (float*)d_out;

  const int bcount = in_sizes[0] / (CCH * HH * WW);   // 16
  dim3 grid((unsigned)(bcount * HH * (WW / XB)));     // 3072 workgroups
  dim3 block(128);                                    // 4 wave32
  hipLaunchKernelGGL(corr81_wmma_kernel, grid, block, 0, stream, f1, f2, out);
}
